// GCNII_53145925321199
// MI455X (gfx1250) — compile-verified, matrix operand order checked by reading, not run on hardware
//
#include <hip/hip_runtime.h>
#include <math.h>

// ---------------------------------------------------------------------------
// GCNII on MI455X (gfx1250), wave32 + WMMA bf16 (bf16x2 split => fp32-class).
// ---------------------------------------------------------------------------

typedef __attribute__((ext_vector_type(16))) __bf16 v16bf;
typedef __attribute__((ext_vector_type(8)))  float  v8f;

#define DIMH     512
#define NNODES   20000
#define NEDGES   320000
#define ALPHA_C  0.1f

// GEMM tiling
#define BM  128
#define BN  128
#define BK  32
#define BKP 40   // padded LDS row stride in bf16 elems (80B: 16B-aligned, conflict-free)

static __device__ __forceinline__ unsigned short bf16_rne(float f) {
  unsigned u = __float_as_uint(f);
  unsigned r = u + 0x7FFFu + ((u >> 16) & 1u);
  return (unsigned short)(r >> 16);
}
static __device__ __forceinline__ float bf16_f32(unsigned short h) {
  return __uint_as_float(((unsigned)h) << 16);
}

union FragU { uint4 q[2]; v16bf v; };

// Fragment load: row-major LDS [row][BKP bf16]; lane needs K = half*8 + (0..7)
// and K = 16 + half*8 + (0..7)  => two ds_load_b128 per fragment.
static __device__ __forceinline__ v16bf load_frag(const unsigned short* lds,
                                                  int row, int laneHalf) {
  FragU f;
  const uint4* p = reinterpret_cast<const uint4*>(lds + row * BKP);
  f.q[0] = p[laneHalf];       // bytes [laneHalf*16, +16)
  f.q[1] = p[laneHalf + 2];   // bytes [32 + laneHalf*16, +16)
  return f.v;
}

static __device__ __forceinline__ void store_hilo4(unsigned short* hiP,
                                                   unsigned short* loP,
                                                   float4 v) {
  unsigned short h0 = bf16_rne(v.x), h1 = bf16_rne(v.y);
  unsigned short h2 = bf16_rne(v.z), h3 = bf16_rne(v.w);
  unsigned short l0 = bf16_rne(v.x - bf16_f32(h0));
  unsigned short l1 = bf16_rne(v.y - bf16_f32(h1));
  unsigned short l2 = bf16_rne(v.z - bf16_f32(h2));
  unsigned short l3 = bf16_rne(v.w - bf16_f32(h3));
  uint2 ph = make_uint2((unsigned)h0 | ((unsigned)h1 << 16),
                        (unsigned)h2 | ((unsigned)h3 << 16));
  uint2 pl = make_uint2((unsigned)l0 | ((unsigned)l1 << 16),
                        (unsigned)l2 | ((unsigned)l3 << 16));
  *reinterpret_cast<uint2*>(hiP) = ph;   // ds_store_b64
  *reinterpret_cast<uint2*>(loP) = pl;
}

// ---------------------------------------------------------------------------
// MODE 0:  Out = A0 @ B0^T + bias            (B0 stored [n][k], e.g. fc_w)
// MODE 1:  acc = (n*(1-a)*A0) @ B0  +  (a*A1) @ B1     (B stored [k][n])
//          Out = relu((1-beta)*(f + f0) + beta*acc + bias)
//          with f = (1-a)*norm[r]*A0, f0 = a*A1
// ---------------------------------------------------------------------------
template <int MODE>
__global__ __launch_bounds__(256) void gcnii_gemm(
    const float* __restrict__ A0, const float* __restrict__ A1,
    const float* __restrict__ B0, const float* __restrict__ B1,
    const float* __restrict__ bias, const float* __restrict__ normv,
    float* __restrict__ Out, float beta, int M) {
  __shared__ __align__(16) unsigned short ldsAhi[BM * BKP];
  __shared__ __align__(16) unsigned short ldsAlo[BM * BKP];
  __shared__ __align__(16) unsigned short ldsBhi[BN * BKP];  // B^T : [n][k]
  __shared__ __align__(16) unsigned short ldsBlo[BN * BKP];

  const int t        = threadIdx.x;
  const int wave     = t >> 5;
  const int lane     = t & 31;
  const int laneHalf = lane >> 4;
  const int laneMod  = lane & 15;
  const int gm0      = blockIdx.y * BM;
  const int gn0      = blockIdx.x * BN;
  const int waveM    = (wave & 3) * 32;  // 4 M-strips of 32 rows
  const int waveN    = (wave >> 2) * 64; // 2 N-strips of 64 cols

  v8f acc[2][4];
#pragma unroll
  for (int i = 0; i < 2; ++i)
#pragma unroll
    for (int j = 0; j < 4; ++j)
#pragma unroll
      for (int e = 0; e < 8; ++e) acc[i][j][e] = 0.0f;

  const int nPhases = (MODE == 1) ? 2 : 1;
  for (int ph = 0; ph < nPhases; ++ph) {
    const float* Ap = (MODE == 1 && ph == 1) ? A1 : A0;
    const float* Bp = (MODE == 1 && ph == 1) ? B1 : B0;

    for (int kc = 0; kc < DIMH; kc += BK) {
      __syncthreads();
      // ---- stage A tile (BM x BK f32 -> bf16 hi/lo), scale folded in ------
#pragma unroll
      for (int i = 0; i < 4; ++i) {
        int q   = t + i * 256;        // 0..1023 float4 slots (128 rows x 8)
        int row = q >> 3;
        int kq  = (q & 7) << 2;
        int gr  = gm0 + row;
        float4 v = make_float4(0.f, 0.f, 0.f, 0.f);
        float scale = 1.0f;
        if (gr < M) {
          v = *reinterpret_cast<const float4*>(Ap + (size_t)gr * DIMH + kc + kq);
          if constexpr (MODE == 1)
            scale = (ph == 0) ? (1.0f - ALPHA_C) * normv[gr] : ALPHA_C;
        }
        v.x *= scale; v.y *= scale; v.z *= scale; v.w *= scale;
        store_hilo4(ldsAhi + row * BKP + kq, ldsAlo + row * BKP + kq, v);
      }
      // ---- stage B tile transposed into [n][k] ----------------------------
#pragma unroll
      for (int i = 0; i < 4; ++i) {
        int q = t + i * 256;
        if constexpr (MODE == 0) {
          // B stored [n][k] already (fc_w): contiguous in k
          int n  = q >> 3;
          int kq = (q & 7) << 2;
          float4 v = *reinterpret_cast<const float4*>(
              Bp + (size_t)(gn0 + n) * DIMH + kc + kq);
          store_hilo4(ldsBhi + n * BKP + kq, ldsBlo + n * BKP + kq, v);
        } else {
          // B stored [k][n]: load float4 along n, scatter to [n][k]
          int k  = q >> 5;
          int nq = (q & 31) << 2;
          float4 v = *reinterpret_cast<const float4*>(
              Bp + (size_t)(kc + k) * DIMH + gn0 + nq);
          float s4[4] = {v.x, v.y, v.z, v.w};
#pragma unroll
          for (int j = 0; j < 4; ++j) {
            unsigned short h = bf16_rne(s4[j]);
            unsigned short l = bf16_rne(s4[j] - bf16_f32(h));
            ldsBhi[(nq + j) * BKP + k] = h;
            ldsBlo[(nq + j) * BKP + k] = l;
          }
        }
      }
      __syncthreads();

      // ---- compute: bf16x2 => 3 WMMAs per (mt,nt) per K-chunk -------------
      v16bf aHi[2], aLo[2];
#pragma unroll
      for (int mt = 0; mt < 2; ++mt) {
        int row = waveM + mt * 16 + laneMod;
        aHi[mt] = load_frag(ldsAhi, row, laneHalf);
        aLo[mt] = load_frag(ldsAlo, row, laneHalf);
      }
#pragma unroll
      for (int nt = 0; nt < 4; ++nt) {
        int col   = waveN + nt * 16 + laneMod;
        v16bf bHi = load_frag(ldsBhi, col, laneHalf);
        v16bf bLo = load_frag(ldsBlo, col, laneHalf);
#pragma unroll
        for (int mt = 0; mt < 2; ++mt) {
          acc[mt][nt] = __builtin_amdgcn_wmma_f32_16x16x32_bf16(
              false, aLo[mt], false, bHi, (short)0, acc[mt][nt], false, false);
          acc[mt][nt] = __builtin_amdgcn_wmma_f32_16x16x32_bf16(
              false, aHi[mt], false, bLo, (short)0, acc[mt][nt], false, false);
          acc[mt][nt] = __builtin_amdgcn_wmma_f32_16x16x32_bf16(
              false, aHi[mt], false, bHi, (short)0, acc[mt][nt], false, false);
        }
      }
    }
  }

  // ---- epilogue: C layout: lane -> N=lane%16, VGPR r -> M=r+(lane/16)*8 ---
#pragma unroll
  for (int mt = 0; mt < 2; ++mt) {
#pragma unroll
    for (int nt = 0; nt < 4; ++nt) {
#pragma unroll
      for (int r = 0; r < 8; ++r) {
        int gm = gm0 + waveM + mt * 16 + laneHalf * 8 + r;
        int gn = gn0 + waveN + nt * 16 + laneMod;
        if (gm < M) {
          float v = acc[mt][nt][r];
          if constexpr (MODE == 0) {
            Out[(size_t)gm * DIMH + gn] = v + bias[gn];
          } else {
            float f   = (1.0f - ALPHA_C) * normv[gm] * A0[(size_t)gm * DIMH + gn];
            float f0  = ALPHA_C * A1[(size_t)gm * DIMH + gn];
            float rst = (1.0f - beta) * (f + f0) + beta * v + bias[gn];
            Out[(size_t)gm * DIMH + gn] = fmaxf(rst, 0.0f);
          }
        }
      }
    }
  }
}

// ---------------------------------------------------------------------------
// Graph kernels (all traffic L2-resident: working set ~120MB < 192MB L2)
// ---------------------------------------------------------------------------
__global__ void zero_kernel(float* p, size_t n) {
  size_t i = (size_t)blockIdx.x * blockDim.x + threadIdx.x;
  size_t s = (size_t)gridDim.x * blockDim.x;
  for (; i < n; i += s) p[i] = 0.0f;
}

__global__ void deg_kernel(const int* __restrict__ dst, float* __restrict__ deg,
                           int E) {
  int i = blockIdx.x * blockDim.x + threadIdx.x;
  if (i < E) unsafeAtomicAdd(&deg[dst[i]], 1.0f);
}

__global__ void norm_kernel(float* nv, int n) {
  int i = blockIdx.x * blockDim.x + threadIdx.x;
  if (i < n) nv[i] = 1.0f / sqrtf(fmaxf(nv[i], 1.0f));
}

// agg[dst[e]][:] += x[src[e]][:] * norm[src[e]] ; one block (128 thr) per edge
__global__ __launch_bounds__(128) void edge_scatter(
    const float* __restrict__ x, const int* __restrict__ src,
    const int* __restrict__ dst, const float* __restrict__ nv,
    float* __restrict__ agg, int E) {
  int e = blockIdx.x;
  int s = src[e];
  int d = dst[e];
  float ns = nv[s];
  int c = threadIdx.x << 2;
  float4 v = *reinterpret_cast<const float4*>(x + (size_t)s * DIMH + c);
  float* out = agg + (size_t)d * DIMH + c;
  unsafeAtomicAdd(out + 0, v.x * ns);
  unsafeAtomicAdd(out + 1, v.y * ns);
  unsafeAtomicAdd(out + 2, v.z * ns);
  unsafeAtomicAdd(out + 3, v.w * ns);
}

// ---------------------------------------------------------------------------
extern "C" void kernel_launch(void* const* d_in, const int* in_sizes, int n_in,
                              void* d_out, int out_size, void* d_ws,
                              size_t ws_size, hipStream_t stream) {
  const float* feat = (const float*)d_in[0];
  const int*   src  = (const int*)d_in[1];
  const int*   dst  = (const int*)d_in[2];
  const float* fc_w = (const float*)d_in[3];
  const float* fc_b = (const float*)d_in[4];
  const float* w1_1 = (const float*)d_in[5];
  const float* w2_1 = (const float*)d_in[6];
  const float* b_1  = (const float*)d_in[7];
  const float* w1_2 = (const float*)d_in[8];
  const float* w2_2 = (const float*)d_in[9];
  const float* b_2  = (const float*)d_in[10];
  float* out = (float*)d_out;

  float* h   = (float*)d_ws;                    // N*D
  float* agg = h + (size_t)NNODES * DIMH;       // N*D
  float* nv  = agg + (size_t)NNODES * DIMH;     // N
  float* res = out;                             // layer-1 output lives in d_out

  const float beta1 = 0.6931471805599453f;  // ln(2)
  const float beta2 = 0.4054651081081644f;  // ln(1.5)

  const dim3 ggrid(DIMH / BN, (NNODES + BM - 1) / BM);  // (4, 157)

  // degree -> norm
  zero_kernel<<<512, 256, 0, stream>>>(nv, (size_t)NNODES);
  deg_kernel<<<(NEDGES + 255) / 256, 256, 0, stream>>>(dst, nv, NEDGES);
  norm_kernel<<<(NNODES + 255) / 256, 256, 0, stream>>>(nv, NNODES);

  // h = feat @ fc_w^T + fc_b
  gcnii_gemm<0><<<ggrid, 256, 0, stream>>>(feat, nullptr, fc_w, nullptr, fc_b,
                                           nullptr, h, 0.0f, NNODES);

  // ---- layer 1: res = relu((1-b1)(f+f0) + b1*(f@w1_1 + f0@w2_1) + b_1) ----
  zero_kernel<<<2048, 256, 0, stream>>>(agg, (size_t)NNODES * DIMH);
  edge_scatter<<<NEDGES, 128, 0, stream>>>(h, src, dst, nv, agg, NEDGES);
  gcnii_gemm<1><<<ggrid, 256, 0, stream>>>(agg, h, w1_1, w2_1, b_1, nv, res,
                                           beta1, NNODES);

  // ---- layer 2 (h0 is still h); final result straight into d_out ---------
  zero_kernel<<<2048, 256, 0, stream>>>(agg, (size_t)NNODES * DIMH);
  edge_scatter<<<NEDGES, 128, 0, stream>>>(res, src, dst, nv, agg, NEDGES);
  gcnii_gemm<1><<<ggrid, 256, 0, stream>>>(agg, h, w1_2, w2_2, b_2, nv, out,
                                           beta2, NNODES);
}
